// Viterbi_23527830848070
// MI455X (gfx1250) — matrix operand order repaired and checked by reading, not compiled, not run on hardware
//
#include <hip/hip_runtime.h>

// ---------------------------------------------------------------------------
// Viterbi-style log-semiring scan:
//   V_new[b,i] = logsumexp_j( theta[t,b,i,j] + V[b,j] ),  t = 0..T-1
//   out[b]     = logsumexp_i V[b,i]
// T=256, B=64, S=128.  theta is 1 GiB fp32, read once -> memory bound
// (23.3 TB/s => ~45us floor).  Strategy: per-batch workgroup, TDM double-
// buffered theta tiles in LDS, exp-space matvec on V_WMMA_F32_16X16X32_F16.
// ---------------------------------------------------------------------------

typedef _Float16     v16h __attribute__((ext_vector_type(16)));
typedef float        v8f  __attribute__((ext_vector_type(8)));
typedef unsigned int v4u  __attribute__((ext_vector_type(4)));
typedef int          v8i  __attribute__((ext_vector_type(8)));
typedef int          v4i  __attribute__((ext_vector_type(4)));

#define TT   256
#define BB   64
#define SS   128
#define SP   129               // padded LDS row stride (dwords) -> no bank conflicts
#define TILE_DW (SS * SS)      // 16384 dwords = 64 KB per step tile
#define BUF_DW (SP * SS)       // 16512 dwords per padded LDS buffer

// Issue one TDM descriptor: contiguous 16384-dword tile from global -> LDS,
// with 1 dword of padding inserted after every 128 dwords (row stride 129).
__device__ __forceinline__ void tdm_load_tile(const float* gptr, unsigned lds_byte_addr) {
    unsigned long long ga = (unsigned long long)(uintptr_t)gptr;
    v4u g0;
    g0[0] = 1u;                                   // count=1, user descriptor
    g0[1] = lds_byte_addr;                        // lds_addr (bytes)
    g0[2] = (unsigned)(ga & 0xFFFFFFFFu);         // global_addr[31:0] -> bits 95:64
    g0[3] = (unsigned)((ga >> 32) & 0x01FFFFFFu)  // global_addr[56:32]
          | (2u << 30);                           // type = 2 ("image")
    v8i g1;
    g1[0] = (2 << 16)      // data_size = 4 bytes
          | (1 << 20)      // pad_enable
          | (6 << 22);     // pad_interval code 6 = 128 dwords
                           // pad_amount code 0 = 1 dword
    g1[1] = (int)((unsigned)TILE_DW << 16);  // tensor_dim0[15:0] @ bits 63:48
    g1[2] = (1 << 16);                       // tensor_dim0[31:16]=0, tensor_dim1=1
    g1[3] = (int)((unsigned)TILE_DW << 16);  // tile_dim0 @ bits 127:112
    g1[4] = 0;                               // tile_dim1 = 0 (1-D tile), tile_dim2 = 0
    g1[5] = TILE_DW;                         // tensor_dim0_stride[31:0]
    g1[6] = 0;
    g1[7] = 0;
    v4i z4 = (v4i)0;
    v8i z8 = (v8i)0;
    // amdgpu-toolchain (clang-23) 6-arg form:
    //   (uint32x4 g0, int32x8 g1, int32x4, int32x4, int32x8, i32 cpol)
    __builtin_amdgcn_tensor_load_to_lds(g0, g1, z4, z4, z8, 0);
}

__global__ void __launch_bounds__(256)
viterbi_scan_kernel(const float* __restrict__ theta, float* __restrict__ out) {
    extern __shared__ float smem[];    // 2 x BUF_DW dwords (double buffer)

    __shared__ float     Vs[SS];       // current state vector (log space)
    __shared__ _Float16  wv[SS];       // exp(V - c), f16
    __shared__ float     cbuf;

    const int tid  = threadIdx.x;
    const int lane = tid & 31;
    const int wave = tid >> 5;          // 0..7, one 16-row tile each
    const int hi   = (lane >= 16) ? 1 : 0;
    const int m    = lane & 15;
    const int b    = blockIdx.x;        // one workgroup per batch

    // LDS byte offset of the double buffer (flat LDS addr low 32 bits = offset)
    const unsigned lds_base = (unsigned)(uintptr_t)(&smem[0]);

    // --- init V0 = [0, -1e10, ..., -1e10], kick off tile t=0 ----------------
    if (tid < SS) Vs[tid] = (tid == 0) ? 0.0f : -1.0e10f;
    if (tid == 0)
        tdm_load_tile(theta + (size_t)b * TILE_DW, lds_base);
    __builtin_amdgcn_s_wait_tensorcnt(0);
    __syncthreads();

    const int rowOff = (wave * 16 + m) * SP;   // A-fragment row base in LDS

    for (int t = 0; t < TT; ++t) {
        const int cur = t & 1, nxt = cur ^ 1;

        // prefetch next tile (overlaps with this step's compute)
        if (tid == 0 && (t + 1) < TT)
            tdm_load_tile(theta + ((size_t)(t + 1) * BB + b) * TILE_DW,
                          lds_base + (unsigned)nxt * (BUF_DW * 4u));

        // c = max_j V[j]   (wave 0 butterfly reduce)
        if (wave == 0) {
            float mx = -3.0e38f;
            #pragma unroll
            for (int k = 0; k < 4; ++k) mx = fmaxf(mx, Vs[lane + 32 * k]);
            #pragma unroll
            for (int off = 16; off; off >>= 1) mx = fmaxf(mx, __shfl_xor(mx, off, 32));
            if (lane == 0) cbuf = mx;
        }
        __syncthreads();
        const float c = cbuf;

        // w_j = exp(V_j - c) in f16  (in [0,1])
        if (tid < SS) wv[tid] = (_Float16)__expf(Vs[tid] - c);
        __syncthreads();

        // s_i = sum_j exp(theta_ij) * w_j  via 4 chained WMMAs (K = 128)
        const float* th = smem + cur * BUF_DW;
        v8f acc = (v8f)0.0f;
        #pragma unroll
        for (int kt = 0; kt < 4; ++kt) {
            const int kbase = kt * 32;
            v16h a, bf;
            #pragma unroll
            for (int r = 0; r < 8; ++r) {
                // A 16x32 f16 layout: K = 16*(r>>2) + 8*hi + 2*(r&3) + h
                const int ka = kbase + ((r & 4) << 2) + (hi ? 8 : 0) + ((r & 3) << 1);
                a[2 * r]     = (_Float16)__expf(th[rowOff + ka]);
                a[2 * r + 1] = (_Float16)__expf(th[rowOff + ka + 1]);
                // B 32x16 f16 layout: K = 16*(r>>2) + 8*hi + 2*(r&3) + h, replicated over N
                const int kb = kbase + ((r & 4) << 2) + (hi ? 8 : 0) + ((r & 3) << 1);
                bf[2 * r]     = wv[kb];
                bf[2 * r + 1] = wv[kb + 1];
            }
            acc = __builtin_amdgcn_wmma_f32_16x16x32_f16(
                    false, a, false, bf, (short)0, acc, false, false);
        }

        // C/D layout: VGPR r holds row M = r + 8*hi, column N = lane&15.
        // All N columns are identical; lanes with N==0 write V_new.
        if (m == 0) {
            #pragma unroll
            for (int r = 0; r < 8; ++r)
                Vs[wave * 16 + (hi ? 8 : 0) + r] = c + __logf(acc[r]);
        }

        __builtin_amdgcn_s_wait_tensorcnt(0);   // next tile landed in LDS
        __syncthreads();                        // V_new + tile visible to all
    }

    // out[b] = logsumexp_i V[i]
    if (wave == 0) {
        float mx = -3.0e38f;
        #pragma unroll
        for (int k = 0; k < 4; ++k) mx = fmaxf(mx, Vs[lane + 32 * k]);
        #pragma unroll
        for (int off = 16; off; off >>= 1) mx = fmaxf(mx, __shfl_xor(mx, off, 32));
        float s = 0.0f;
        #pragma unroll
        for (int k = 0; k < 4; ++k) s += __expf(Vs[lane + 32 * k] - mx);
        #pragma unroll
        for (int off = 16; off; off >>= 1) s += __shfl_xor(s, off, 32);
        if (lane == 0) out[b] = mx + __logf(s);
    }
}

extern "C" void kernel_launch(void* const* d_in, const int* in_sizes, int n_in,
                              void* d_out, int out_size, void* d_ws, size_t ws_size,
                              hipStream_t stream) {
    (void)in_sizes; (void)n_in; (void)out_size; (void)d_ws; (void)ws_size;
    const float* theta = (const float*)d_in[0];
    float* out = (float*)d_out;
    const size_t shmem = 2u * BUF_DW * sizeof(float);   // 132,096 B double buffer
    viterbi_scan_kernel<<<dim3(BB), dim3(256), shmem, stream>>>(theta, out);
}
